// SelfAttention_39487929319510
// MI455X (gfx1250) — matrix operand order repaired and checked by reading, not compiled, not run on hardware
//
#include <hip/hip_runtime.h>
#include <math.h>

// ---------------- problem constants ----------------
#define B_    4
#define SEQ_  1024
#define DIM_  1024
#define H_    16
#define DH_   64
#define DHP_  128           // padded head dim for attn@V GEMM (N multiple of 128)
#define KV_   2304          // CMEM(256) + MEM(1024) + SEQ(1024)
#define NK2_  4608          // KV_ * 2 (qk | qpe columns concatenated)
#define TMEM_ 1280          // total_mem = cmem + mem
#define CLEN_ 256
#define SCALE_ 0.125f       // DH^-0.5
#define LN_EPS_ 1e-5f

typedef __attribute__((ext_vector_type(16))) __bf16 v16bf;
typedef __attribute__((ext_vector_type(8)))  float  v8f;
typedef __attribute__((ext_vector_type(8)))  int    v8i;

__device__ __forceinline__ unsigned short f2bf(float f) {
    unsigned int u = __float_as_uint(f);
    u = (u + 0x7FFFu + ((u >> 16) & 1u)) >> 16;   // round-to-nearest-even
    return (unsigned short)u;
}

// ======================================================================
// Generic batched GEMM:  C[M,N] (f32) = A[M,K] (bf16, row-major, lda)
//                                     @ B[K,N] given as BT[N,K] (bf16, ldb)
// REQUIREMENT: M % 128 == 0, N % 128 == 0, K % 32 == 0 (all call sites comply)
// grid = (N/128, M/128, batch), block = 256 (8 waves)
// wave tile: 32(M) x 64(N)  -> 2x4 WMMA 16x16x32 accumulators
// Double-buffered LDS, filled with GLOBAL_LOAD_ASYNC_TO_LDS_B128 (ASYNCcnt).
// ======================================================================
#define BM 128
#define BN 128
#define BK 32
#define LDSP 40   // padded row stride (u16): 80B rows, keeps 16B alignment

__global__ __launch_bounds__(256)
void gemm_bf16_wmma(const unsigned short* __restrict__ A, long long sA, int lda,
                    const unsigned short* __restrict__ BT, long long sB, int ldb,
                    float* __restrict__ C, long long sC, int ldc,
                    int M, int N, int K)
{
    __shared__ unsigned short As[2][BM][LDSP];
    __shared__ unsigned short Bs[2][BN][LDSP];

    A  += (long long)blockIdx.z * sA;
    BT += (long long)blockIdx.z * sB;
    C  += (long long)blockIdx.z * sC;

    const int tid   = threadIdx.x;
    const int lane  = tid & 31;
    const int wave  = tid >> 5;          // 0..7
    const int waveM = wave >> 1;         // 0..3 (32 rows each)
    const int waveN = wave & 1;          // 0..1 (64 cols each)
    const int mlo   = lane & 15;
    const int hi    = (lane >> 4) & 1;

    const int block_m = blockIdx.y * BM;
    const int block_n = blockIdx.x * BN;

    const unsigned long long Abase = (unsigned long long)A;
    const unsigned long long Bbase = (unsigned long long)BT;

    // per-thread staging chunks: chunk c -> row = c>>2, col8 = (c&3)*8 u16
    const int r0 = tid >> 2,          c0 = (tid & 3) << 3;
    const int r1 = (tid + 256) >> 2,  c1 = c0;            // same col group

    // async-stage one BK-tile of A and BT into LDS buffer `buf`
    auto stage = [&](int buf, int k0) {
        {
            unsigned lofA = (unsigned)(unsigned long long)&As[buf][r0][c0];
            unsigned gofA = (unsigned)(((long long)(block_m + r0) * lda + k0 + c0) * 2);
            asm volatile("global_load_async_to_lds_b128 %0, %1, %2"
                         :: "v"(lofA), "v"(gofA), "s"(Abase) : "memory");
            unsigned lofB = (unsigned)(unsigned long long)&Bs[buf][r0][c0];
            unsigned gofB = (unsigned)(((long long)(block_n + r0) * ldb + k0 + c0) * 2);
            asm volatile("global_load_async_to_lds_b128 %0, %1, %2"
                         :: "v"(lofB), "v"(gofB), "s"(Bbase) : "memory");
        }
        {
            unsigned lofA = (unsigned)(unsigned long long)&As[buf][r1][c1];
            unsigned gofA = (unsigned)(((long long)(block_m + r1) * lda + k0 + c1) * 2);
            asm volatile("global_load_async_to_lds_b128 %0, %1, %2"
                         :: "v"(lofA), "v"(gofA), "s"(Abase) : "memory");
            unsigned lofB = (unsigned)(unsigned long long)&Bs[buf][r1][c1];
            unsigned gofB = (unsigned)(((long long)(block_n + r1) * ldb + k0 + c1) * 2);
            asm volatile("global_load_async_to_lds_b128 %0, %1, %2"
                         :: "v"(lofB), "v"(gofB), "s"(Bbase) : "memory");
        }
    };

    v8f zero = {0.f,0.f,0.f,0.f,0.f,0.f,0.f,0.f};
    v8f acc[2][4];
#pragma unroll
    for (int i = 0; i < 2; ++i)
#pragma unroll
        for (int j = 0; j < 4; ++j) acc[i][j] = zero;

    stage(0, 0);
    asm volatile("s_wait_asynccnt 0x0" ::: "memory");
    __syncthreads();

    const int nk = K / BK;
    int buf = 0;
    for (int kt = 0; kt < nk; ++kt) {
        if (kt + 1 < nk) stage(buf ^ 1, (kt + 1) * BK);

        // ---- fragments per ISA layout ----
        v8i afr[2], bfr[4];
#pragma unroll
        for (int tm = 0; tm < 2; ++tm) {
            int mrow = waveM * 32 + tm * 16 + mlo;
#pragma unroll
            for (int r = 0; r < 8; ++r) {
                int kc = ((r >> 2) << 4) + (hi << 3) + ((r & 3) << 1);
                afr[tm][r] = *(const int*)(&As[buf][mrow][kc]);
            }
        }
#pragma unroll
        for (int tn = 0; tn < 4; ++tn) {
            int nrow = waveN * 64 + tn * 16 + mlo;
#pragma unroll
            for (int r = 0; r < 8; ++r) {
                int kc = ((r >> 2) << 4) + (hi << 3) + ((r & 3) << 1);
                bfr[tn][r] = *(const int*)(&Bs[buf][nrow][kc]);
            }
        }
#pragma unroll
        for (int tm = 0; tm < 2; ++tm)
#pragma unroll
            for (int tn = 0; tn < 4; ++tn)
                acc[tm][tn] = __builtin_amdgcn_wmma_f32_16x16x32_bf16(
                    false, __builtin_bit_cast(v16bf, afr[tm]),
                    false, __builtin_bit_cast(v16bf, bfr[tn]),
                    (short)0, acc[tm][tn], false, false);

        asm volatile("s_wait_asynccnt 0x0" ::: "memory");
        __syncthreads();
        buf ^= 1;
    }

    // ---- store: C/D layout m = 8*(lane/16)+r, n = lane%16 ----
#pragma unroll
    for (int tm = 0; tm < 2; ++tm)
#pragma unroll
        for (int tn = 0; tn < 4; ++tn)
#pragma unroll
            for (int r = 0; r < 8; ++r) {
                int m = block_m + waveM * 32 + tm * 16 + hi * 8 + r;
                int n = block_n + waveN * 64 + tn * 16 + mlo;
                C[(long long)m * ldc + n] = acc[tm][tn][r];
            }
}

// ======================================================================
// elementwise / pack kernels
// ======================================================================
__global__ void transpose_convert(const float* __restrict__ W,
                                  unsigned short* __restrict__ out,
                                  int K, int N, long long cnt)   // out[n*K+k] = W[k*N+n]
{
    long long i = (long long)blockIdx.x * 256 + threadIdx.x;
    if (i >= cnt) return;
    int n = (int)(i / K);
    int k = (int)(i % K);
    out[i] = f2bf(W[(long long)k * N + n]);
}

__global__ void pack_convw(const float* __restrict__ w, unsigned short* __restrict__ out)
{   // conv_w (OC,IC,4) -> WcT[oc][r*1024+ic]
    long long i = (long long)blockIdx.x * 256 + threadIdx.x;
    if (i >= (long long)DIM_ * 4096) return;
    int oc  = (int)(i >> 12);
    int rem = (int)(i & 4095);
    int r   = rem >> 10;
    int ic  = rem & 1023;
    out[i]  = f2bf(w[(long long)oc * 4096 + ic * 4 + r]);
}

__global__ void build_kvin(const float* __restrict__ cm, const float* __restrict__ mm,
                           const float* __restrict__ xx, unsigned short* __restrict__ out)
{
    long long i = (long long)blockIdx.x * 256 + threadIdx.x;
    if (i >= (long long)B_ * KV_ * DIM_) return;
    int e = (int)(i & 1023);
    long long t = i >> 10;
    int j = (int)(t % KV_);
    int b = (int)(t / KV_);
    float v;
    if (j < CLEN_)      v = cm[((long long)b * CLEN_ + j) * DIM_ + e];
    else if (j < TMEM_) v = mm[((long long)b * SEQ_ + (j - CLEN_)) * DIM_ + e];
    else                v = xx[((long long)b * SEQ_ + (j - TMEM_)) * DIM_ + e];
    out[i] = f2bf(v);
}

__global__ void pack_q(const float* __restrict__ Q, unsigned short* __restrict__ out)
{   // Qf32[b][m][h*64+d] -> qbf[b][h][m][d]
    long long i = (long long)blockIdx.x * 256 + threadIdx.x;
    if (i >= (long long)B_ * H_ * SEQ_ * DH_) return;
    int d  = (int)(i & 63);
    int m  = (int)((i >> 6) & 1023);
    int bh = (int)(i >> 16);
    int b  = bh >> 4, h = bh & 15;
    out[i] = f2bf(Q[(((long long)b << 10) + m) * DIM_ + (h << 6) + d]);
}

__global__ void pack_k(const float* __restrict__ KVf, unsigned short* __restrict__ kpeT)
{   // K half of KV -> kpeT[b][h][j][d]   (rows 0..KV-1 of the NK2 block)
    long long i = (long long)blockIdx.x * 256 + threadIdx.x;
    if (i >= (long long)B_ * H_ * KV_ * DH_) return;
    int d  = (int)(i & 63);
    long long t = i >> 6;
    int j  = (int)(t % KV_);
    int bh = (int)(t / KV_);
    int b  = bh >> 4, h = bh & 15;
    kpeT[(((long long)bh * NK2_) + j) * DH_ + d] =
        f2bf(KVf[(((long long)b * KV_) + j) * (2 * DIM_) + (h << 6) + d]);
}

__global__ void pack_pe(const float* __restrict__ pe, unsigned short* __restrict__ kpeT)
{   // pos_embed[h][j][d] -> kpeT rows KV..NK2-1 (duplicated per b)
    long long i = (long long)blockIdx.x * 256 + threadIdx.x;
    if (i >= (long long)B_ * H_ * KV_ * DH_) return;
    int d  = (int)(i & 63);
    long long t = i >> 6;
    int j  = (int)(t % KV_);
    int bh = (int)(t / KV_);
    int h  = bh & 15;
    kpeT[(((long long)bh * NK2_) + KV_ + j) * DH_ + d] =
        f2bf(pe[((long long)h * KV_ + j) * DH_ + d]);
}

__global__ void pack_v(const float* __restrict__ KVf, unsigned short* __restrict__ vT)
{   // V half of KV -> vT[b][h][d][j], padded to DHP_=128 rows (rows 64..127 = 0)
    long long i = (long long)blockIdx.x * 256 + threadIdx.x;
    if (i >= (long long)B_ * H_ * DHP_ * KV_) return;
    int j  = (int)(i % KV_);
    long long t = i / KV_;
    int d  = (int)(t & 127);
    int bh = (int)(t >> 7);
    int b  = bh >> 4, h = bh & 15;
    unsigned short v = 0;
    if (d < DH_)
        v = f2bf(KVf[(((long long)b * KV_) + j) * (2 * DIM_) + DIM_ + (h << 6) + d]);
    vT[i] = v;
}

__global__ void reorder_out(const float* __restrict__ aout, unsigned short* __restrict__ outbf)
{   // aout[b][h][m][0..DHP) -> outbf[b][m][h*64+d]  (d < 64)
    long long i = (long long)blockIdx.x * 256 + threadIdx.x;
    if (i >= (long long)B_ * SEQ_ * DIM_) return;
    int e  = (int)(i & 1023);
    int h  = e >> 6, d = e & 63;
    long long bm = i >> 10;
    int m = (int)(bm & 1023);
    int b = (int)(bm >> 10);
    outbf[i] = f2bf(aout[((((long long)b * H_ + h) * SEQ_) + m) * DHP_ + d]);
}

__global__ void copy_mem(const float* __restrict__ x, float* __restrict__ dst)
{
    long long i = (long long)blockIdx.x * 256 + threadIdx.x;
    if (i < (long long)B_ * SEQ_ * DIM_) dst[i] = x[i];
}

__global__ void conv_bias(const float* __restrict__ cb, float* __restrict__ dst)
{
    long long i = (long long)blockIdx.x * 256 + threadIdx.x;
    if (i < (long long)B_ * CLEN_ * DIM_) dst[i] += cb[i & 1023];
}

__global__ void set_aux(float* __restrict__ dst) { dst[0] = 0.0f; }

// ======================================================================
// fused scale + relative-shift + mask + softmax + bf16 requantize
// scores layout per b: [H][SEQ][NK2]  (cols 0..KV-1 = q·k, KV.. = q·pe)
// shift: add pos[m, n + SEQ-1 - m] (0 if index >= KV)
// ======================================================================
__global__ __launch_bounds__(256)
void softmax_kernel(const float* __restrict__ scores,
                    const unsigned char* __restrict__ imask,   // [SEQ] for this b
                    unsigned short* __restrict__ attn)          // [H][SEQ][KV]
{
    const int m = blockIdx.x;
    const int h = blockIdx.y;
    const int tid = threadIdx.x;
    const float* qk = scores + (((long long)h * SEQ_) + m) * NK2_;
    const float* pp = qk + KV_;
    unsigned short* out = attn + (((long long)h * SEQ_) + m) * KV_;

    __shared__ float sv[KV_];
    __shared__ float red[256];

    const bool mi = imask[m] != 0;
    float maxv = -3.402823466e38f;
    for (int n = tid; n < KV_; n += 256) {
        int idx = n + (SEQ_ - 1) - m;
        float pe = (idx < KV_) ? pp[idx] : 0.0f;
        float val = (qk[n] + pe) * SCALE_;
        bool masked = (n > m + TMEM_);
        if (n >= TMEM_ && !(mi && (imask[n - TMEM_] != 0))) masked = true;
        val = masked ? -3.402823466e38f : val;
        sv[n] = val;
        maxv = fmaxf(maxv, val);
    }
    red[tid] = maxv; __syncthreads();
    for (int s = 128; s > 0; s >>= 1) { if (tid < s) red[tid] = fmaxf(red[tid], red[tid + s]); __syncthreads(); }
    maxv = red[0]; __syncthreads();

    float sum = 0.0f;
    for (int n = tid; n < KV_; n += 256) {
        float e = __expf(sv[n] - maxv);
        sv[n] = e; sum += e;
    }
    red[tid] = sum; __syncthreads();
    for (int s = 128; s > 0; s >>= 1) { if (tid < s) red[tid] += red[tid + s]; __syncthreads(); }
    float inv = 1.0f / red[0];
    for (int n = tid; n < KV_; n += 256) out[n] = f2bf(sv[n] * inv);
}

// ======================================================================
// residual + bias + LayerNorm,  one block per (b, m) row
// ======================================================================
__global__ __launch_bounds__(256)
void ln_residual(const float* __restrict__ logits, const float* __restrict__ x,
                 const float* __restrict__ bout, const float* __restrict__ g,
                 const float* __restrict__ bb, float* __restrict__ out)
{
    const long long row = blockIdx.x;
    const float* lg = logits + row * DIM_;
    const float* xr = x + row * DIM_;
    float* o = out + row * DIM_;
    const int tid = threadIdx.x;
    __shared__ float y[DIM_];
    __shared__ float red[256];

    float s = 0.0f;
    for (int e = tid; e < DIM_; e += 256) {
        float v = xr[e] + lg[e] + bout[e];
        y[e] = v; s += v;
    }
    red[tid] = s; __syncthreads();
    for (int t = 128; t > 0; t >>= 1) { if (tid < t) red[tid] += red[tid + t]; __syncthreads(); }
    float mu = red[0] * (1.0f / DIM_); __syncthreads();

    float s2 = 0.0f;
    for (int e = tid; e < DIM_; e += 256) { float d = y[e] - mu; s2 += d * d; }
    red[tid] = s2; __syncthreads();
    for (int t = 128; t > 0; t >>= 1) { if (tid < t) red[tid] += red[tid + t]; __syncthreads(); }
    float var = red[0] * (1.0f / DIM_);
    float inv = rsqrtf(var + LN_EPS_);
    for (int e = tid; e < DIM_; e += 256) o[e] = (y[e] - mu) * inv * g[e] + bb[e];
}

// ======================================================================
extern "C" void kernel_launch(void* const* d_in, const int* in_sizes, int n_in,
                              void* d_out, int out_size, void* d_ws, size_t ws_size,
                              hipStream_t stream)
{
    (void)in_sizes; (void)n_in; (void)out_size; (void)ws_size;
    const float* x     = (const float*)d_in[0];
    const float* mem   = (const float*)d_in[1];
    const float* cmem  = (const float*)d_in[2];
    const float* pe    = (const float*)d_in[3];
    const unsigned char* imask = (const unsigned char*)d_in[4];
    const float* Wq    = (const float*)d_in[5];
    const float* Wkv   = (const float*)d_in[6];
    const float* Wout  = (const float*)d_in[7];
    const float* bout  = (const float*)d_in[8];
    const float* lng   = (const float*)d_in[9];
    const float* lnb   = (const float*)d_in[10];
    const float* convw = (const float*)d_in[11];
    const float* convb = (const float*)d_in[12];
    float* out = (float*)d_out;

    char* p = (char*)d_ws;
    auto alloc = [&](size_t bytes) -> void* {
        void* r = (void*)p;
        p += (bytes + 255) & ~(size_t)255;
        return r;
    };

    unsigned short* kvin  = (unsigned short*)alloc((size_t)B_ * KV_ * DIM_ * 2);
    unsigned short* WqT   = (unsigned short*)alloc((size_t)DIM_ * DIM_ * 2);
    unsigned short* WkvT  = (unsigned short*)alloc((size_t)2 * DIM_ * DIM_ * 2);
    unsigned short* WoutT = (unsigned short*)alloc((size_t)DIM_ * DIM_ * 2);
    unsigned short* WcT   = (unsigned short*)alloc((size_t)DIM_ * 4096 * 2);
    float* Qf             = (float*)alloc((size_t)B_ * SEQ_ * DIM_ * 4);
    float* KVf            = (float*)alloc((size_t)B_ * KV_ * 2 * DIM_ * 4);
    unsigned short* qbf   = (unsigned short*)alloc((size_t)B_ * H_ * SEQ_ * DH_ * 2);
    unsigned short* kpeT  = (unsigned short*)alloc((size_t)B_ * H_ * NK2_ * DH_ * 2);
    unsigned short* vT    = (unsigned short*)alloc((size_t)B_ * H_ * DHP_ * KV_ * 2);
    float* scores         = (float*)alloc((size_t)H_ * SEQ_ * NK2_ * 4);         // per-b reuse
    unsigned short* attnbf= (unsigned short*)alloc((size_t)H_ * SEQ_ * KV_ * 2); // per-b reuse
    float* aout           = (float*)alloc((size_t)B_ * H_ * SEQ_ * DHP_ * 4);
    unsigned short* outbf = (unsigned short*)alloc((size_t)B_ * SEQ_ * DIM_ * 2);
    float* logitsf        = (float*)alloc((size_t)B_ * SEQ_ * DIM_ * 4);

    const int T = 256;
    auto nb = [](long long n) { return (unsigned)((n + 255) / 256); };

    // ---- weight conversions ----
    transpose_convert<<<nb((long long)DIM_ * DIM_), T, 0, stream>>>(Wq, WqT, DIM_, DIM_, (long long)DIM_ * DIM_);
    transpose_convert<<<nb((long long)2 * DIM_ * DIM_), T, 0, stream>>>(Wkv, WkvT, DIM_, 2 * DIM_, (long long)2 * DIM_ * DIM_);
    transpose_convert<<<nb((long long)DIM_ * DIM_), T, 0, stream>>>(Wout, WoutT, DIM_, DIM_, (long long)DIM_ * DIM_);
    pack_convw<<<nb((long long)DIM_ * 4096), T, 0, stream>>>(convw, WcT);
    build_kvin<<<nb((long long)B_ * KV_ * DIM_), T, 0, stream>>>(cmem, mem, x, kvin);

    // ---- Q = x @ Wq   (A = x rows inside kvin) ----
    gemm_bf16_wmma<<<dim3(8, 8, B_), T, 0, stream>>>(
        kvin + (size_t)TMEM_ * DIM_, (long long)KV_ * DIM_, DIM_,
        WqT, 0, DIM_,
        Qf, (long long)SEQ_ * DIM_, DIM_,
        SEQ_, DIM_, DIM_);

    // ---- KV = kv_input @ Wkv ----
    gemm_bf16_wmma<<<dim3(16, 18, B_), T, 0, stream>>>(
        kvin, (long long)KV_ * DIM_, DIM_,
        WkvT, 0, DIM_,
        KVf, (long long)KV_ * 2 * DIM_, 2 * DIM_,
        KV_, 2 * DIM_, DIM_);

    // ---- head packing ----
    pack_q <<<nb((long long)B_ * H_ * SEQ_ * DH_), T, 0, stream>>>(Qf, qbf);
    pack_k <<<nb((long long)B_ * H_ * KV_ * DH_), T, 0, stream>>>(KVf, kpeT);
    pack_pe<<<nb((long long)B_ * H_ * KV_ * DH_), T, 0, stream>>>(pe, kpeT);
    pack_v <<<nb((long long)B_ * H_ * DHP_ * KV_), T, 0, stream>>>(KVf, vT);

    // ---- attention, per batch (workspace reuse) ----
    for (int b = 0; b < B_; ++b) {
        // scores[h][m][0:KV) = q·k ; [KV:NK2) = q·pe   (batched over heads)
        gemm_bf16_wmma<<<dim3(36, 8, H_), T, 0, stream>>>(
            qbf + (size_t)b * H_ * SEQ_ * DH_, (long long)SEQ_ * DH_, DH_,
            kpeT + (size_t)b * H_ * NK2_ * DH_, (long long)NK2_ * DH_, DH_,
            scores, (long long)SEQ_ * NK2_, NK2_,
            SEQ_, NK2_, DH_);

        softmax_kernel<<<dim3(SEQ_, H_), T, 0, stream>>>(scores, imask + (size_t)b * SEQ_, attnbf);

        // out = attn @ v   (N padded to 128; rows 64..127 of vT are zero)
        gemm_bf16_wmma<<<dim3(1, 8, H_), T, 0, stream>>>(
            attnbf, (long long)SEQ_ * KV_, KV_,
            vT + (size_t)b * H_ * DHP_ * KV_, (long long)DHP_ * KV_, KV_,
            aout + (size_t)b * H_ * SEQ_ * DHP_, (long long)SEQ_ * DHP_, DHP_,
            SEQ_, DHP_, KV_);
    }

    // ---- merge heads + output projection ----
    reorder_out<<<nb((long long)B_ * SEQ_ * DIM_), T, 0, stream>>>(aout, outbf);
    gemm_bf16_wmma<<<dim3(8, 32, 1), T, 0, stream>>>(
        outbf, 0, DIM_,
        WoutT, 0, DIM_,
        logitsf, 0, DIM_,
        B_ * SEQ_, DIM_, DIM_);

    // ---- residual + LayerNorm -> logits output ----
    ln_residual<<<dim3(B_ * SEQ_), T, 0, stream>>>(logitsf, x, bout, lng, lnb, out);

    // ---- new_mem = x ----
    copy_mem<<<nb((long long)B_ * SEQ_ * DIM_), T, 0, stream>>>(x, out + 4194304);

    // ---- new_cmem = conv1d(mem) as GEMM: A = mem rows inside kvin viewed [256][4096] ----
    gemm_bf16_wmma<<<dim3(8, 2, B_), T, 0, stream>>>(
        kvin + (size_t)CLEN_ * DIM_, (long long)KV_ * DIM_, 4096,
        WcT, 0, 4096,
        out + 8388608, (long long)CLEN_ * DIM_, DIM_,
        CLEN_, DIM_, 4096);
    conv_bias<<<nb((long long)B_ * CLEN_ * DIM_), T, 0, stream>>>(convb, out + 8388608);

    // ---- aux_loss = 0 ----
    set_aux<<<1, 1, 0, stream>>>(out + 9437184);
}
